// MoE_5299989643304
// MI455X (gfx1250) — matrix operand order repaired and checked by reading, not compiled, not run on hardware
//
#include <hip/hip_runtime.h>
#include <hip/hip_bf16.h>

// ---------------- problem constants (from setup_inputs) ----------------
#define T_TOK   8192        // B*L tokens
#define D_DIM   1024        // model dim
#define H_DIM   512         // expert hidden dim
#define E_NUM   16          // experts
#define TOPK    2
#define TILE_M  32          // tokens per workgroup tile
#define MAX_TILES (T_TOK / TILE_M)   // 256 (worst case: all tokens -> one expert)

typedef __attribute__((ext_vector_type(16))) __bf16 bf16x16;
typedef __attribute__((ext_vector_type(8)))  float  f32x8;

union Frag32B { uint4 q[2]; bf16x16 v; };   // 32 bytes = one WMMA A/B fragment per lane

__device__ __forceinline__ unsigned short f32_to_bf16(float f) {
  unsigned int u = __builtin_bit_cast(unsigned int, f);
  u += 0x7FFFu + ((u >> 16) & 1u);          // round-to-nearest-even
  return (unsigned short)(u >> 16);
}

// ---------------- kernel 1: zero output + expert counters ----------------
__global__ void zero_kernel(float* __restrict__ y, int n, int* __restrict__ counts) {
  int gid = blockIdx.x * blockDim.x + threadIdx.x;
  if (gid < n) y[gid] = 0.0f;
  if (gid < E_NUM) counts[gid] = 0;
}

// ---------------- kernel 2: f32 -> bf16 convert (+ weight transpose) ----------------
// xbf:  [T][D]       (row-major, as x)
// W1T:  [E][H][D]    (W1T[e][n][k] = W1[e][k][n])  -> B fragments are K-contiguous
// W2T:  [E][D][H]    (W2T[e][n][k] = W2[e][k][n])
__global__ void convert_kernel(const float* __restrict__ x,
                               const float* __restrict__ W1,
                               const float* __restrict__ W2,
                               unsigned short* __restrict__ xbf,
                               unsigned short* __restrict__ W1T,
                               unsigned short* __restrict__ W2T) {
  long gid = (long)blockIdx.x * blockDim.x + threadIdx.x;
  const long NX = (long)T_TOK * D_DIM;
  const long NW = (long)E_NUM * D_DIM * H_DIM;
  if (gid < NX) { xbf[gid] = f32_to_bf16(x[gid]); return; }
  gid -= NX;
  if (gid < NW) {
    int e   = (int)(gid / ((long)H_DIM * D_DIM));
    int rem = (int)(gid - (long)e * H_DIM * D_DIM);
    int n = rem / D_DIM, k = rem % D_DIM;
    W1T[(long)e * H_DIM * D_DIM + rem] =
        f32_to_bf16(W1[((long)e * D_DIM + k) * H_DIM + n]);
    return;
  }
  gid -= NW;
  if (gid < NW) {
    int e   = (int)(gid / ((long)D_DIM * H_DIM));
    int rem = (int)(gid - (long)e * D_DIM * H_DIM);
    int n = rem / H_DIM, k = rem % H_DIM;
    W2T[(long)e * D_DIM * H_DIM + rem] =
        f32_to_bf16(W2[((long)e * H_DIM + k) * D_DIM + n]);
  }
}

// ---------------- kernel 3: top-2 softmax gating + expert bucketing ----------------
// One wave32 per token. 16 accumulators per lane, butterfly reduction.
__global__ __launch_bounds__(128) void gate_kernel(const float* __restrict__ x,
                                                   const float* __restrict__ wg,
                                                   int*   __restrict__ counts,
                                                   int*   __restrict__ etok,
                                                   float* __restrict__ egate) {
  int wave = threadIdx.x >> 5;
  int lane = threadIdx.x & 31;
  int t = blockIdx.x * 4 + wave;
  if (t >= T_TOK) return;
  const float* xr = x + (long)t * D_DIM;

  float acc[E_NUM];
#pragma unroll
  for (int e = 0; e < E_NUM; ++e) acc[e] = 0.0f;

  for (int d = lane; d < D_DIM; d += 32) {          // x reads fully coalesced
    float xv = xr[d];
    const float4* wr = (const float4*)(wg + (long)d * E_NUM);
    float4 w0 = wr[0], w1 = wr[1], w2 = wr[2], w3 = wr[3];  // 64B row, L2 resident
    acc[0]  += xv * w0.x;  acc[1]  += xv * w0.y;  acc[2]  += xv * w0.z;  acc[3]  += xv * w0.w;
    acc[4]  += xv * w1.x;  acc[5]  += xv * w1.y;  acc[6]  += xv * w1.z;  acc[7]  += xv * w1.w;
    acc[8]  += xv * w2.x;  acc[9]  += xv * w2.y;  acc[10] += xv * w2.z;  acc[11] += xv * w2.w;
    acc[12] += xv * w3.x;  acc[13] += xv * w3.y;  acc[14] += xv * w3.z;  acc[15] += xv * w3.w;
  }
#pragma unroll
  for (int off = 16; off > 0; off >>= 1) {
#pragma unroll
    for (int e = 0; e < E_NUM; ++e) acc[e] += __shfl_xor(acc[e], off, 32);
  }
  if (lane == 0) {
    float mx = acc[0];
#pragma unroll
    for (int e = 1; e < E_NUM; ++e) mx = fmaxf(mx, acc[e]);
    float p[E_NUM], s = 0.0f;
#pragma unroll
    for (int e = 0; e < E_NUM; ++e) { p[e] = __expf(acc[e] - mx); s += p[e]; }
    float inv = 1.0f / s;
    int i0 = 0; float v0 = p[0];
#pragma unroll
    for (int e = 1; e < E_NUM; ++e) if (p[e] > v0) { v0 = p[e]; i0 = e; }
    int i1 = -1; float v1 = -1.0f;
#pragma unroll
    for (int e = 0; e < E_NUM; ++e) if (e != i0 && p[e] > v1) { v1 = p[e]; i1 = e; }
    int p0 = atomicAdd(&counts[i0], 1);
    etok[i0 * T_TOK + p0]  = t;
    egate[i0 * T_TOK + p0] = v0 * inv;   // gates are raw softmax probs (no renorm)
    int p1 = atomicAdd(&counts[i1], 1);
    etok[i1 * T_TOK + p1]  = t;
    egate[i1 * T_TOK + p1] = v1 * inv;
  }
}

// ---------------- kernel 4: per-expert fused GEMM1 + gelu + GEMM2 + scatter ----------------
// Block = 256 threads = 8 waves; one (expert, 32-token tile) per block.
// Wave w: M-subtile = w&1 (16 tokens), N-group = w>>1.
__global__ __launch_bounds__(256) void moe_gemm_kernel(
    const unsigned short* __restrict__ xbf,
    const unsigned short* __restrict__ W1T,
    const unsigned short* __restrict__ W2T,
    const int*   __restrict__ counts,
    const int*   __restrict__ etok,
    const float* __restrict__ egate,
    float* __restrict__ y) {
  __shared__ __align__(16) unsigned short hlds[TILE_M * H_DIM]; // 32 KB bf16 h-tile
  __shared__ int   stok[TILE_M];
  __shared__ float sgate[TILE_M];

  int e    = blockIdx.x / MAX_TILES;
  int tile = blockIdx.x % MAX_TILES;
  int cnt  = counts[e];
  if (tile * TILE_M >= cnt) return;          // uniform per block (before any barrier)

  int tid = threadIdx.x;
  if (tid < TILE_M) {
    int idx = tile * TILE_M + tid;
    if (idx < cnt) { stok[tid] = etok[e * T_TOK + idx]; sgate[tid] = egate[e * T_TOK + idx]; }
    else           { stok[tid] = 0;                      sgate[tid] = 0.0f; }  // pad: gate 0 => adds 0
  }
  __syncthreads();

  int wave = tid >> 5, lane = tid & 31;
  int msub = wave & 1, ngroup = wave >> 1;
  int half = lane >> 4, nl = lane & 15;
  int mbase = msub * 16;
  long arow = (long)stok[mbase + nl] * D_DIM;     // A-fragment row (M = nl per ISA layout)

  // -------- GEMM1: h[32,512] = X[32,1024] @ W1[e]  (bf16 WMMA, f32 acc) --------
  f32x8 acc[8];
#pragma unroll
  for (int i = 0; i < 8; ++i) acc[i] = (f32x8)0.0f;

  for (int kb = 0; kb < D_DIM; kb += 32) {
    Frag32B a;                                    // A 16x32 bf16: K = half*8..+7 and 16+half*8..+7
    const unsigned short* ap = xbf + arow + kb + half * 8;
    a.q[0] = *(const uint4*)(ap);
    a.q[1] = *(const uint4*)(ap + 16);
#pragma unroll
    for (int i = 0; i < 8; ++i) {
      int ncol = (ngroup * 8 + i) * 16 + nl;      // output column (B's N = nl per lane)
      const unsigned short* bp = W1T + ((long)e * H_DIM + ncol) * D_DIM + kb + half * 8;
      Frag32B b;                                  // B 32x16: K = half*8..+7 and 16+half*8..+7
      b.q[0] = *(const uint4*)(bp);
      b.q[1] = *(const uint4*)(bp + 16);
      acc[i] = __builtin_amdgcn_wmma_f32_16x16x32_bf16(
          false, a.v, false, b.v, (short)0, acc[i], false, false);
    }
  }
  // exact gelu, convert to bf16, stash h tile in LDS (C layout: M = r + 8*half, N = nl)
#pragma unroll
  for (int i = 0; i < 8; ++i) {
    int ncol = (ngroup * 8 + i) * 16 + nl;
#pragma unroll
    for (int r = 0; r < 8; ++r) {
      int m = mbase + half * 8 + r;
      float v = acc[i][r];
      float g = 0.5f * v * (1.0f + erff(v * 0.70710678118f));
      hlds[m * H_DIM + ncol] = f32_to_bf16(g);
    }
  }
  __syncthreads();

  // -------- GEMM2: y[32,1024] += gate * (h[32,512] @ W2[e]) --------
  const unsigned short* hrow = hlds + (mbase + nl) * H_DIM;   // A row for this lane
  for (int pass = 0; pass < 2; ++pass) {          // 2 passes x 8 N-tiles = 16 tiles/wave
    f32x8 c[8];
#pragma unroll
    for (int i = 0; i < 8; ++i) c[i] = (f32x8)0.0f;

    for (int kb = 0; kb < H_DIM; kb += 32) {
      Frag32B a;
      const unsigned short* ap = hrow + kb + half * 8;        // ds_load_b128 x2
      a.q[0] = *(const uint4*)(ap);
      a.q[1] = *(const uint4*)(ap + 16);
#pragma unroll
      for (int i = 0; i < 8; ++i) {
        int ncol = (ngroup * 16 + pass * 8 + i) * 16 + nl;
        const unsigned short* bp = W2T + ((long)e * D_DIM + ncol) * H_DIM + kb + half * 8;
        Frag32B b;
        b.q[0] = *(const uint4*)(bp);
        b.q[1] = *(const uint4*)(bp + 16);
        c[i] = __builtin_amdgcn_wmma_f32_16x16x32_bf16(
            false, a.v, false, b.v, (short)0, c[i], false, false);
      }
    }
    // gate-scaled scatter-add (k=2 experts may hit same token row -> atomic f32)
#pragma unroll
    for (int i = 0; i < 8; ++i) {
      int ncol = (ngroup * 16 + pass * 8 + i) * 16 + nl;
#pragma unroll
      for (int r = 0; r < 8; ++r) {
        int m = mbase + half * 8 + r;
        float val = c[i][r] * sgate[m];
        atomicAdd(&y[(long)stok[m] * D_DIM + ncol], val);
      }
    }
  }
}

// ---------------- host-side launcher ----------------
extern "C" void kernel_launch(void* const* d_in, const int* in_sizes, int n_in,
                              void* d_out, int out_size, void* d_ws, size_t ws_size,
                              hipStream_t stream) {
  const float* x  = (const float*)d_in[0];   // [T, D]
  const float* wg = (const float*)d_in[1];   // [D, E]
  const float* W1 = (const float*)d_in[2];   // [E, D, H]
  const float* W2 = (const float*)d_in[3];   // [E, H, D]
  float* y = (float*)d_out;                  // [T, D]

  // workspace layout (~51.4 MB)
  unsigned short* xbf = (unsigned short*)d_ws;                         // 16 MB
  unsigned short* W1T = xbf + (long)T_TOK * D_DIM;                     // 16 MB
  unsigned short* W2T = W1T + (long)E_NUM * H_DIM * D_DIM;             // 16 MB
  int*   counts = (int*)(W2T + (long)E_NUM * D_DIM * H_DIM);           // 64 B (padded)
  int*   etok   = counts + 64;                                         // 512 KB
  float* egate  = (float*)(etok + E_NUM * T_TOK);                      // 512 KB

  int nY = T_TOK * D_DIM;
  zero_kernel<<<(nY + 255) / 256, 256, 0, stream>>>(y, nY, counts);

  long nConv = (long)T_TOK * D_DIM + 2L * E_NUM * D_DIM * H_DIM;
  convert_kernel<<<(int)((nConv + 255) / 256), 256, 0, stream>>>(x, W1, W2, xbf, W1T, W2T);

  gate_kernel<<<T_TOK / 4, 128, 0, stream>>>(x, wg, counts, etok, egate);

  moe_gemm_kernel<<<E_NUM * MAX_TILES, 256, 0, stream>>>(xbf, W1T, W2T,
                                                         counts, etok, egate, y);
}
